// UnifiedAttention_18399639896804
// MI455X (gfx1250) — matrix operand order repaired and checked by reading, not compiled, not run on hardware
//
#include <hip/hip_runtime.h>
#include <hip/hip_bf16.h>

typedef __attribute__((ext_vector_type(16))) __bf16 v16bf;
typedef __attribute__((ext_vector_type(8)))  __bf16 v8bf;
typedef __attribute__((ext_vector_type(8)))  float  v8f;

#define WMMA_BF16(a, b, c) \
  __builtin_amdgcn_wmma_f32_16x16x32_bf16(false, (a), false, (b), (short)0, (c), false, false)

__device__ __forceinline__ __bf16 f2bf(float f) {
  unsigned u = __float_as_uint(f);
  u += 0x7FFFu + ((u >> 16) & 1u);   // round-to-nearest-even
  unsigned short s = (unsigned short)(u >> 16);
  union { unsigned short u16; __bf16 b; } cv;
  cv.u16 = s;
  return cv.b;
}

// A-fragment (16x32 bf16, MxK): lane holds row M=lane%16; K chunk at
// koff..koff+15 (koff already includes 16*(lane>>4)).
__device__ __forceinline__ v16bf load_a_frag(const __bf16* row, int koff) {
  v8bf lo = *(const v8bf*)(row + koff);
  v8bf hi = *(const v8bf*)(row + koff + 8);
  v16bf a;
#pragma unroll
  for (int i = 0; i < 8; ++i) { a[i] = lo[i]; a[i + 8] = hi[i]; }
  return a;
}

// ---------------------------------------------------------------- convert
__global__ __launch_bounds__(256) void cvt_bf16_kernel(const float* __restrict__ src,
                                                       __bf16* __restrict__ dst, int n) {
  int i = blockIdx.x * blockDim.x + threadIdx.x;
  int stride = gridDim.x * blockDim.x;
  for (; i < n; i += stride) dst[i] = f2bf(src[i]);
}

// ------------------------------------------------------- QKV fused GEMM
// out[m, j] = sum_k Xb[m,k] * Wb[j,k] + bias[j];  M=4096 (B*N), J=3072, K=1024.
// Q: scaled by 1/8, layout [B,H,N,64]; K: [B,H,N,64]; V: transposed [B,H,64,N].
__global__ __launch_bounds__(256) void qkv_gemm_kernel(
    const __bf16* __restrict__ Xb, const __bf16* __restrict__ Wb,
    const float* __restrict__ bias,
    __bf16* __restrict__ Qb, __bf16* __restrict__ Kb, __bf16* __restrict__ Vt) {
  const int lane = threadIdx.x & 31;
  const int wave = threadIdx.x >> 5;
  const int half = lane >> 4;
  const int l16  = lane & 15;
  const int jt = blockIdx.x * 64;                 // 48 col groups
  const int mt = (blockIdx.y * 8 + wave) * 16;    // 256 row tiles
  const int K = 1024;

  const __bf16* arow = Xb + (size_t)(mt + l16) * K;
  v8f acc[4] = {};
  for (int kb = 0; kb < K; kb += 32) {
    v16bf a = load_a_frag(arow, kb + 16 * half);
#pragma unroll
    for (int ct = 0; ct < 4; ++ct) {
      v16bf bm = *(const v16bf*)(Wb + (size_t)(jt + ct * 16 + l16) * K + kb + 16 * half);
      acc[ct] = WMMA_BF16(a, bm, acc[ct]);
    }
  }
  const int s = jt >> 10;            // 0=Q,1=K,2=V (uniform per wave tile)
  const int h = (jt >> 6) & 15;
  const int bidx = mt >> 11;
  const int n0 = mt & 2047;
#pragma unroll
  for (int ct = 0; ct < 4; ++ct) {
    const int j = jt + ct * 16 + l16;
    const int d = j & 63;
    const float bj = bias[j];
    if (s == 2) {                    // V transposed: packed 16B store along n
      v8bf pk;
#pragma unroll
      for (int r = 0; r < 8; ++r) pk[r] = f2bf(acc[ct][r] + bj);
      size_t off = (((size_t)bidx * 16 + h) * 64 + d) * 2048 + (size_t)(n0 + 8 * half);
      *(v8bf*)(Vt + off) = pk;
    } else {
      __bf16* dst = (s == 0) ? Qb : Kb;
      const float sc = (s == 0) ? 0.125f : 1.0f;   // fold attn scale into Q
#pragma unroll
      for (int r = 0; r < 8; ++r) {
        int n = n0 + r + 8 * half;
        dst[(((size_t)bidx * 16 + h) * 2048 + n) * 64 + d] = f2bf((acc[ct][r] + bj) * sc);
      }
    }
  }
}

// ------------------------------------------------------ flash attention
// Computes S^T = K.Q^T so each lane owns one query column; online softmax
// is lane-local except one shfl_xor(16). O^T = V^T.P^T; P relaid via LDS.
__global__ __launch_bounds__(256) void flash_attn_kernel(
    const __bf16* __restrict__ Qb, const __bf16* __restrict__ Kb,
    const __bf16* __restrict__ Vt, const int* __restrict__ amask,
    __bf16* __restrict__ AO) {
  __shared__ __attribute__((aligned(64))) __bf16 Plds[8][16][32];
  const int lane = threadIdx.x & 31;
  const int wave = threadIdx.x >> 5;
  const int half = lane >> 4;
  const int l16  = lane & 15;
  const int bh = blockIdx.x;               // B*H = 32
  const int b = bh >> 4;
  const int h = bh & 15;
  const int N = 2048;
  const int qbase = blockIdx.y * 128 + wave * 16;
  const int q = qbase + l16;               // this lane's query

  const __bf16* Qrow = Qb + ((size_t)bh * N + q) * 64;
  const v16bf bq0 = *(const v16bf*)(Qrow + 16 * half);        // d 0..31
  const v16bf bq1 = *(const v16bf*)(Qrow + 32 + 16 * half);   // d 32..63

  const __bf16* Kbase = Kb + (size_t)bh * N * 64;
  const __bf16* Vbase = Vt + (size_t)bh * 64 * N;
  const int* mrow = amask + ((size_t)b * N + q) * N;

  v8f o[4] = {};
  float m_i = -__builtin_inff();
  float l_i = 0.0f;

  const int kend = qbase + 15;             // causal: skip fully-masked chunks
  for (int kb = 0; kb <= kend; kb += 32) {
    v8f st[2];
#pragma unroll
    for (int t = 0; t < 2; ++t) {
      const __bf16* krow = Kbase + (size_t)(kb + 16 * t + l16) * 64;
      v16bf ak0 = load_a_frag(krow, 16 * half);
      v16bf ak1 = load_a_frag(krow, 32 + 16 * half);
      v8f sacc = {};
      sacc = WMMA_BF16(ak0, bq0, sacc);
      sacc = WMMA_BF16(ak1, bq1, sacc);
      st[t] = sacc;
    }
    float sv[16];
    float mloc = m_i;
#pragma unroll
    for (int t = 0; t < 2; ++t) {
      const int key0 = kb + 16 * t + 8 * half;
      const int4 mk0 = *(const int4*)(mrow + key0);
      const int4 mk1 = *(const int4*)(mrow + key0 + 4);
      const int mkv[8] = {mk0.x, mk0.y, mk0.z, mk0.w, mk1.x, mk1.y, mk1.z, mk1.w};
#pragma unroll
      for (int r = 0; r < 8; ++r) {
        float sc = st[t][r];
        const int key = key0 + r;
        if (key > q || mkv[r] == 0) sc = -1.0e9f;   // causal + padding mask
        sv[t * 8 + r] = sc;
        mloc = fmaxf(mloc, sc);
      }
    }
    const float m_new = fmaxf(mloc, __shfl_xor(mloc, 16, 32));
    const float alpha = __expf(m_i - m_new);
    m_i = m_new;
    l_i *= alpha;
#pragma unroll
    for (int u = 0; u < 4; ++u) o[u] = o[u] * alpha;
#pragma unroll
    for (int t = 0; t < 2; ++t) {          // P^T (C-layout) -> LDS [q][key]
      v8bf p;
#pragma unroll
      for (int r = 0; r < 8; ++r) {
        const float e = __expf(sv[t * 8 + r] - m_new);
        l_i += e;
        p[r] = f2bf(e);
      }
      *(v8bf*)&Plds[wave][l16][16 * t + 8 * half] = p;
    }
    const v16bf bp = *(const v16bf*)&Plds[wave][l16][16 * half];  // B-layout reload
#pragma unroll
    for (int u = 0; u < 4; ++u) {          // O^T += V^T (16d x 32k) . P^T
      const __bf16* vrow = Vbase + (size_t)(u * 16 + l16) * N + kb;
      v16bf av = load_a_frag(vrow, 16 * half);
      o[u] = WMMA_BF16(av, bp, o[u]);
    }
  }
  const float l_tot = l_i + __shfl_xor(l_i, 16, 32);
  const float inv = (l_tot > 0.0f) ? (1.0f / l_tot) : 0.0f;
  const size_t row = (size_t)b * N + q;
#pragma unroll
  for (int u = 0; u < 4; ++u) {            // packed 16B stores along d
    v8bf pk;
#pragma unroll
    for (int r = 0; r < 8; ++r) pk[r] = f2bf(o[u][r] * inv);
    *(v8bf*)(AO + row * 1024 + h * 64 + u * 16 + 8 * half) = pk;
  }
}

// ------------------------------------------------------------ proj GEMM
__global__ __launch_bounds__(256) void proj_gemm_kernel(
    const __bf16* __restrict__ Ab, const __bf16* __restrict__ Wb,
    const float* __restrict__ bias, float* __restrict__ out) {
  const int lane = threadIdx.x & 31;
  const int wave = threadIdx.x >> 5;
  const int half = lane >> 4;
  const int l16  = lane & 15;
  const int jt = blockIdx.x * 64;
  const int mt = (blockIdx.y * 8 + wave) * 16;
  const int K = 1024;
  const __bf16* arow = Ab + (size_t)(mt + l16) * K;
  v8f acc[4] = {};
  for (int kb = 0; kb < K; kb += 32) {
    v16bf a = load_a_frag(arow, kb + 16 * half);
#pragma unroll
    for (int ct = 0; ct < 4; ++ct) {
      v16bf bm = *(const v16bf*)(Wb + (size_t)(jt + ct * 16 + l16) * K + kb + 16 * half);
      acc[ct] = WMMA_BF16(a, bm, acc[ct]);
    }
  }
#pragma unroll
  for (int ct = 0; ct < 4; ++ct) {
    const int j = jt + ct * 16 + l16;
    const float bj = bias[j];
#pragma unroll
    for (int r = 0; r < 8; ++r)
      out[(size_t)(mt + r + 8 * half) * 1024 + j] = acc[ct][r] + bj;
  }
}

// ---------------------------------------------------------------- launch
extern "C" void kernel_launch(void* const* d_in, const int* in_sizes, int n_in,
                              void* d_out, int out_size, void* d_ws, size_t ws_size,
                              hipStream_t stream) {
  const float* x      = (const float*)d_in[0];
  const int*   amask  = (const int*)d_in[1];
  const float* qkv_w  = (const float*)d_in[2];
  const float* qkv_b  = (const float*)d_in[3];
  const float* proj_w = (const float*)d_in[4];
  const float* proj_b = (const float*)d_in[5];
  float* out = (float*)d_out;

  char* ws = (char*)d_ws;
  size_t off = 0;
  auto alloc = [&](size_t bytes) -> char* {
    char* p = ws + off;
    off += (bytes + 255) & ~(size_t)255;
    return p;
  };
  __bf16* Xb  = (__bf16*)alloc((size_t)4096 * 1024 * 2);           // x bf16
  __bf16* Wqb = (__bf16*)alloc((size_t)3072 * 1024 * 2);           // qkv_w bf16
  __bf16* Wpb = (__bf16*)alloc((size_t)1024 * 1024 * 2);           // proj_w bf16
  __bf16* Qb  = (__bf16*)alloc((size_t)2 * 16 * 2048 * 64 * 2);    // [B,H,N,D]
  __bf16* Kb  = (__bf16*)alloc((size_t)2 * 16 * 2048 * 64 * 2);    // [B,H,N,D]
  __bf16* Vt  = (__bf16*)alloc((size_t)2 * 16 * 64 * 2048 * 2);    // [B,H,D,N]
  __bf16* AO  = (__bf16*)alloc((size_t)4096 * 1024 * 2);           // attn out

  cvt_bf16_kernel<<<1024, 256, 0, stream>>>(x, Xb, 4096 * 1024);
  cvt_bf16_kernel<<<1024, 256, 0, stream>>>(qkv_w, Wqb, 3072 * 1024);
  cvt_bf16_kernel<<<512, 256, 0, stream>>>(proj_w, Wpb, 1024 * 1024);
  qkv_gemm_kernel<<<dim3(48, 32), 256, 0, stream>>>(Xb, Wqb, qkv_b, Qb, Kb, Vt);
  flash_attn_kernel<<<dim3(32, 16), 256, 0, stream>>>(Qb, Kb, Vt, amask, AO);
  proj_gemm_kernel<<<dim3(16, 32), 256, 0, stream>>>(AO, Wpb, proj_b, out);
}
